// xLSTMBlock_7224134992406
// MI455X (gfx1250) — matrix operand order repaired and checked by reading, not compiled, not run on hardware
//
#include <hip/hip_runtime.h>
#include <hip/hip_bf16.h>
#include <math.h>

#define B_  2
#define S_  2048
#define D_  1024
#define NH_ 8
#define DH_ 128

#define BM_ 128          // rows per workgroup (8 waves x 16 rows)
#define BN_ 32           // columns (j) per tile

#define KPAD 136         // LDS stride (halves) for K tile rows
#define VTPAD 40         // LDS stride (halves) for V^T rows (per d)
#define PPAD 40          // LDS stride (halves) for per-wave P buffer rows

#define NEGINF (-1e30f)
#define RSCALE 0.08838834764831845f   // 1/sqrt(128)
#define EPS_  1e-6f
#define LNEPS 1e-5f

typedef __attribute__((ext_vector_type(16))) _Float16 v16h;
typedef __attribute__((ext_vector_type(8)))  float    v8f;

union Frag16 { v16h v; _Float16 h[16]; unsigned int u[8]; };
union Pack4  { _Float16 h[4]; unsigned long long u; };

__device__ __forceinline__ v8f zero8() {
    v8f z;
#pragma unroll
    for (int i = 0; i < 8; ++i) z[i] = 0.0f;
    return z;
}

// XOR-lane shuffle via ds_swizzle_b32 immediate (group-of-32 mode):
// offset[15]=0, xor_mask=offset[14:10], or_mask=offset[9:5]=0, and_mask=offset[4:0]=0x1f
template <int XM>
__device__ __forceinline__ float swz_xor(float x) {
    const int y = __builtin_amdgcn_ds_swizzle(__float_as_int(x), (XM << 10) | 0x1f);
    return __int_as_float(y);
}
template <int XM>
__device__ __forceinline__ float red_max16(float x) {   // max over 16-lane half
    return fmaxf(x, swz_xor<XM>(x));
}

// ---------------------------------------------------------------------------
// Kernel 1: gate projections.  ig = ifg @ igw.T + igb ; log_f = logsigmoid(fg)
// One block per (b,s).  Outputs: aout[(b*NH+h)*S+s] = ig (later becomes a),
//                                lfout[...]         = log_sigmoid(fg) (later cs)
// ---------------------------------------------------------------------------
__global__ __launch_bounds__(256)
void gate_kernel(const float* __restrict__ q, const float* __restrict__ k,
                 const float* __restrict__ v,
                 const float* __restrict__ igw, const float* __restrict__ igb,
                 const float* __restrict__ fgw, const float* __restrict__ fgb,
                 float* __restrict__ aout, float* __restrict__ lfout)
{
    __shared__ float red[16][257];
    const int bs = blockIdx.x;            // b*S + s
    const int b  = bs / S_;
    const int s  = bs % S_;
    const int tid = threadIdx.x;

    const float* qr = q + (size_t)bs * D_;
    const float* kr = k + (size_t)bs * D_;
    const float* vr = v + (size_t)bs * D_;

    float accv[16];
#pragma unroll
    for (int o = 0; o < 16; ++o) accv[o] = 0.0f;

    for (int d = tid; d < D_; d += 256) {
        const float qv = qr[d], kv = kr[d], vv = vr[d];
#pragma unroll
        for (int hh = 0; hh < 8; ++hh) {
            const float* wi = igw + (size_t)hh * 3 * D_;
            accv[hh]     += qv * wi[d] + kv * wi[D_ + d] + vv * wi[2 * D_ + d];
            const float* wf = fgw + (size_t)hh * 3 * D_;
            accv[8 + hh] += qv * wf[d] + kv * wf[D_ + d] + vv * wf[2 * D_ + d];
        }
    }
#pragma unroll
    for (int o = 0; o < 16; ++o) red[o][tid] = accv[o];
    __syncthreads();

    if (tid < 16) {
        float sum = 0.0f;
        for (int i = 0; i < 256; ++i) sum += red[tid][i];
        const int hh = tid & 7;
        const size_t idx = ((size_t)b * NH_ + hh) * S_ + s;
        if (tid < 8) {
            aout[idx] = sum + igb[hh];
        } else {
            const float fgp = sum + fgb[hh];
            // log_sigmoid(x) = min(x,0) - log1p(exp(-|x|))
            const float lf = fminf(fgp, 0.0f) - log1pf(__expf(-fabsf(fgp)));
            lfout[idx] = lf;
        }
    }
}

// ---------------------------------------------------------------------------
// Kernel 2: per-(b,h) inclusive cumsum of log_f -> cs; a = ig - cs (in place)
// ---------------------------------------------------------------------------
__global__ __launch_bounds__(256)
void scan_kernel(float* __restrict__ csbuf, float* __restrict__ abuf)
{
    __shared__ float buf[S_];
    __shared__ float part[257];
    const int bh = blockIdx.x;
    const size_t base = (size_t)bh * S_;
    const int tid = threadIdx.x;

    for (int i = tid; i < S_; i += 256) buf[i] = csbuf[base + i];
    __syncthreads();

    float tot = 0.0f;
#pragma unroll
    for (int j = 0; j < 8; ++j) {
        const int i = tid * 8 + j;
        tot += buf[i];
        buf[i] = tot;
    }
    part[tid] = tot;
    __syncthreads();
    if (tid == 0) {
        float r = 0.0f;
        for (int i = 0; i < 256; ++i) { const float t = part[i]; part[i] = r; r += t; }
    }
    __syncthreads();
    const float off = part[tid];
#pragma unroll
    for (int j = 0; j < 8; ++j) buf[tid * 8 + j] += off;
    __syncthreads();

    for (int i = tid; i < S_; i += 256) {
        const float c = buf[i];
        csbuf[base + i] = c;
        abuf[base + i]  = abuf[base + i] - c;   // a[j] = ig[j] - cs[j]
    }
}

// ---------------------------------------------------------------------------
// Kernel 3: fused causal mLSTM attention with WMMA.
//   - log_D[i,j] = cs_i + a_j  =>  decay weight exp(a_j - A_i) is column-only;
//     full tiles use one uniform running max A (single diagonal tile per wave
//     is masked per row).
//   - row sums computed by an extra WMMA against an all-ones B fragment.
//   - double-buffered LDS tiles: next tile's global loads issue before the
//     compute block; one barrier per iteration.
//   - B fragments staged in half-batches of 4 (32 VGPRs) to stay under the
//     256-VGPR boundary; lane reductions use ds_swizzle_b32 immediates.
// Grid: B*NH*(S/BM_), Block: 256 (8 wave32, 16 rows each).
// ---------------------------------------------------------------------------
__global__ __launch_bounds__(256)
void mlstm_attn_kernel(const float* __restrict__ q, const float* __restrict__ k,
                       const float* __restrict__ v,
                       const float* __restrict__ csg, const float* __restrict__ ag,
                       float* __restrict__ hout)
{
    __shared__ _Float16 ldsK[2][32 * KPAD];     // K tile row-major [j][kfeat]
    __shared__ _Float16 ldsVT[2][DH_ * VTPAD];  // V tile transposed [d][j]
    __shared__ _Float16 ldsP[8 * 16 * PPAD];    // per-wave P buffer [row][j]

    const int blk = blockIdx.x;
    const int tiles = S_ / BM_;                 // 16
    const int b   = blk / (NH_ * tiles);
    const int rem = blk % (NH_ * tiles);
    const int h   = rem / tiles;
    const int rt  = rem % tiles;
    const int row0 = rt * BM_;

    const int tid  = threadIdx.x;
    const int wave = tid >> 5;
    const int lane = tid & 31;
    const int ln16 = lane & 15;
    const int hi   = lane >> 4;                 // 0 / 1
    const int half8 = hi * 8;

    const int wrow0 = row0 + wave * 16;

    // per-thread slice of the cooperative tile load (4 float4 per matrix)
    const int ldr[4]  = { (tid + 0*256) >> 5, (tid + 1*256) >> 5,
                          (tid + 2*256) >> 5, (tid + 3*256) >> 5 };
    const int ldc4[4] = { ((tid + 0*256) & 31) * 4, ((tid + 1*256) & 31) * 4,
                          ((tid + 2*256) & 31) * 4, ((tid + 3*256) & 31) * 4 };

    // ---- load Q fragments for this wave's 16-row strip (A layout) ----
    Frag16 qf[4];
    {
        const int qm = wrow0 + ln16;
        const float* qrow = q + ((size_t)(b * S_ + qm)) * D_ + h * DH_;
#pragma unroll
        for (int kc = 0; kc < 4; ++kc) {
#pragma unroll
            for (int g = 0; g < 4; ++g) {
                // half-pairs are contiguous in groups of 4 floats
                const int k0 = kc * 32 + (g < 2 ? hi * 8 + 4 * g
                                                : 16 + hi * 8 + 4 * (g - 2));
                const float4 f = *(const float4*)(qrow + k0);
                qf[kc].h[4 * g + 0] = (_Float16)f.x;
                qf[kc].h[4 * g + 1] = (_Float16)f.y;
                qf[kc].h[4 * g + 2] = (_Float16)f.z;
                qf[kc].h[4 * g + 3] = (_Float16)f.w;
            }
        }
    }

    // ---- all-ones B fragment for WMMA row sums ----
    Frag16 ones;
#pragma unroll
    for (int i = 0; i < 16; ++i) ones.h[i] = (_Float16)1.0f;

    // ---- running stats: per-row max of valid a_j; row sums in accs ----
    const float* abh = ag + ((size_t)(b * NH_ + h)) * S_;
    float Arow[8];
#pragma unroll
    for (int r = 0; r < 8; ++r) Arow[r] = NEGINF;
    v8f acc[8];
#pragma unroll
    for (int t = 0; t < 8; ++t) acc[t] = zero8();
    v8f accs = zero8();

    const float* kbh = k + ((size_t)b * S_) * D_ + h * DH_;
    const float* vbh = v + ((size_t)b * S_) * D_ + h * DH_;
    _Float16* pb = ldsP + wave * 16 * PPAD;

    // ---- prologue: fill buffer 0 with tile 0 ----
    {
#pragma unroll
        for (int ii = 0; ii < 4; ++ii) {
            const int r = ldr[ii], c4 = ldc4[ii];
            const float4 k4 = *(const float4*)(kbh + (size_t)r * D_ + c4);
            const float4 v4 = *(const float4*)(vbh + (size_t)r * D_ + c4);
            Pack4 pk;
            pk.h[0] = (_Float16)k4.x; pk.h[1] = (_Float16)k4.y;
            pk.h[2] = (_Float16)k4.z; pk.h[3] = (_Float16)k4.w;
            *(unsigned long long*)&ldsK[0][r * KPAD + c4] = pk.u;
            ldsVT[0][(c4 + 0) * VTPAD + r] = (_Float16)v4.x;
            ldsVT[0][(c4 + 1) * VTPAD + r] = (_Float16)v4.y;
            ldsVT[0][(c4 + 2) * VTPAD + r] = (_Float16)v4.z;
            ldsVT[0][(c4 + 3) * VTPAD + r] = (_Float16)v4.w;
        }
    }
    __syncthreads();

    const int jtiles = (row0 + BM_) / BN_;
    for (int jt = 0; jt < jtiles; ++jt) {
        const int jbase = jt * BN_;
        const int cur = jt & 1;
        const int nxt = cur ^ 1;
        const bool have_next = (jt + 1 < jtiles);

        // ---- issue next tile's global loads into registers (no wait) ----
        float4 kreg[4], vreg[4];
        if (have_next) {
            const int jb2 = jbase + BN_;
#pragma unroll
            for (int ii = 0; ii < 4; ++ii) {
                kreg[ii] = *(const float4*)(kbh + (size_t)(jb2 + ldr[ii]) * D_ + ldc4[ii]);
                vreg[ii] = *(const float4*)(vbh + (size_t)(jb2 + ldr[ii]) * D_ + ldc4[ii]);
            }
        }

        // ---- compute on current buffer ----
        if (jbase <= wrow0 + 15) {          // wave-uniform causal skip
            const _Float16* Kc = ldsK[cur];
            const _Float16* Vc = ldsVT[cur];

            // QK^T: stage K fragments in half-batches of 4, 4-deep WMMA chains
            v8f sc0 = zero8(), sc1 = zero8();
#pragma unroll
            for (int hb = 0; hb < 2; ++hb) {
                Frag16 kb[4];
#pragma unroll
                for (int kc2 = 0; kc2 < 2; ++kc2) {
                    const int kc = hb * 2 + kc2;
#pragma unroll
                    for (int v8 = 0; v8 < 8; ++v8) {
                        const int kk = kc * 32 + (v8 < 4 ? hi * 8 + 2 * v8 : 16 + hi * 8 + 2 * (v8 - 4));
                        kb[2 * kc2].u[v8]     = *(const unsigned int*)&Kc[ln16 * KPAD + kk];
                        kb[2 * kc2 + 1].u[v8] = *(const unsigned int*)&Kc[(16 + ln16) * KPAD + kk];
                    }
                }
#pragma unroll
                for (int kc2 = 0; kc2 < 2; ++kc2) {
                    const int kc = hb * 2 + kc2;
                    sc0 = __builtin_amdgcn_wmma_f32_16x16x32_f16(false, qf[kc].v, false, kb[2 * kc2].v,
                                                                 (short)0, sc0, false, false);
                    sc1 = __builtin_amdgcn_wmma_f32_16x16x32_f16(false, qf[kc].v, false, kb[2 * kc2 + 1].v,
                                                                 (short)0, sc1, false, false);
                }
            }

            const float aj0 = abh[jbase + ln16];
            const float aj1 = abh[jbase + 16 + ln16];

            if (jbase + 31 <= wrow0) {
                // ===== fast path: tile fully causal-valid for all 16 rows =====
                float am = fmaxf(aj0, aj1);
                am = red_max16<1>(am);
                am = red_max16<2>(am);
                am = red_max16<4>(am);
                am = red_max16<8>(am);
                const float Aold  = Arow[0];       // rows identical in fast phase
                const float Anew  = fmaxf(Aold, am);
                const float scale = __expf(Aold - Anew);
                const float w0 = RSCALE * __expf(aj0 - Anew);
                const float w1 = RSCALE * __expf(aj1 - Anew);
#pragma unroll
                for (int r = 0; r < 8; ++r) {
                    Arow[r] = Anew;
                    accs[r] *= scale;
#pragma unroll
                    for (int t = 0; t < 8; ++t) acc[t][r] *= scale;
                    const float p0 = sc0[r] * w0;
                    const float p1 = sc1[r] * w1;
                    pb[(r + half8) * PPAD + ln16]      = (_Float16)p0;
                    pb[(r + half8) * PPAD + 16 + ln16] = (_Float16)p1;
                }
            } else {
                // ===== diagonal tile (once per wave): per-row masking =====
                const int j0 = jbase + ln16;
                const int j1 = jbase + 16 + ln16;
#pragma unroll
                for (int r = 0; r < 8; ++r) {
                    const int mrow = wrow0 + r + half8;
                    const bool v0 = (j0 <= mrow);
                    const bool v1 = (j1 <= mrow);
                    float l0 = v0 ? aj0 : NEGINF;
                    float l1 = v1 ? aj1 : NEGINF;
                    float loc = fmaxf(l0, l1);
                    loc = red_max16<1>(loc);
                    loc = red_max16<2>(loc);
                    loc = red_max16<4>(loc);
                    loc = red_max16<8>(loc);
                    const float Anew  = fmaxf(Arow[r], loc);
                    const float scale = __expf(Arow[r] - Anew);
                    const float p0 = v0 ? sc0[r] * RSCALE * __expf(aj0 - Anew) : 0.0f;
                    const float p1 = v1 ? sc1[r] * RSCALE * __expf(aj1 - Anew) : 0.0f;
                    Arow[r] = Anew;
                    accs[r] *= scale;
#pragma unroll
                    for (int t = 0; t < 8; ++t) acc[t][r] *= scale;
                    pb[(r + half8) * PPAD + ln16]      = (_Float16)p0;
                    pb[(r + half8) * PPAD + 16 + ln16] = (_Float16)p1;
                }
            }

            // P (A layout) from per-wave LDS buffer
            Frag16 pf;
#pragma unroll
            for (int v8 = 0; v8 < 8; ++v8) {
                const int k0 = (v8 < 4 ? hi * 8 + 2 * v8 : 16 + hi * 8 + 2 * (v8 - 4));
                pf.u[v8] = *(const unsigned int*)&pb[ln16 * PPAD + k0];
            }
            // P @ V: stage V fragments in half-batches of 4, 4-deep WMMA chains
#pragma unroll
            for (int g = 0; g < 2; ++g) {
                Frag16 vbf[4];
#pragma unroll
                for (int t4 = 0; t4 < 4; ++t4) {
                    const int t = g * 4 + t4;
#pragma unroll
                    for (int v8 = 0; v8 < 8; ++v8) {
                        const int kk = (v8 < 4 ? hi * 8 + 2 * v8 : 16 + hi * 8 + 2 * (v8 - 4));         // B layout (K = j)
                        vbf[t4].u[v8] = *(const unsigned int*)&Vc[(t * 16 + ln16) * VTPAD + kk];
                    }
                }
#pragma unroll
                for (int t4 = 0; t4 < 4; ++t4) {
                    const int t = g * 4 + t4;
                    acc[t] = __builtin_amdgcn_wmma_f32_16x16x32_f16(false, pf.v, false, vbf[t4].v,
                                                                    (short)0, acc[t], false, false);
                }
            }
            // row sums of P via WMMA against all-ones B
            accs = __builtin_amdgcn_wmma_f32_16x16x32_f16(false, pf.v, false, ones.v,
                                                          (short)0, accs, false, false);
        }

        // ---- convert + store next tile into the other buffer ----
        if (have_next) {
            _Float16* Kn = ldsK[nxt];
            _Float16* Vn = ldsVT[nxt];
#pragma unroll
            for (int ii = 0; ii < 4; ++ii) {
                const int r = ldr[ii], c4 = ldc4[ii];
                Pack4 pk;
                pk.h[0] = (_Float16)kreg[ii].x; pk.h[1] = (_Float16)kreg[ii].y;
                pk.h[2] = (_Float16)kreg[ii].z; pk.h[3] = (_Float16)kreg[ii].w;
                *(unsigned long long*)&Kn[r * KPAD + c4] = pk.u;
                Vn[(c4 + 0) * VTPAD + r] = (_Float16)vreg[ii].x;
                Vn[(c4 + 1) * VTPAD + r] = (_Float16)vreg[ii].y;
                Vn[(c4 + 2) * VTPAD + r] = (_Float16)vreg[ii].z;
                Vn[(c4 + 3) * VTPAD + r] = (_Float16)vreg[ii].w;
            }
        }
        __syncthreads();
    }

    // ---- epilogue: normalizer + store h ----
    const float* csbh = csg + ((size_t)(b * NH_ + h)) * S_;
    float inv[8];
#pragma unroll
    for (int r = 0; r < 8; ++r) {
        const float csm  = csbh[wrow0 + r + half8];
        const float norm = fmaxf(fabsf(accs[r]), __expf(-(csm + Arow[r]))) + EPS_;
        inv[r] = 1.0f / norm;
    }
#pragma unroll
    for (int t = 0; t < 8; ++t) {
#pragma unroll
        for (int r = 0; r < 8; ++r) {
            const int mrow = wrow0 + r + half8;
            hout[((size_t)(b * S_ + mrow)) * D_ + h * DH_ + t * 16 + ln16] = acc[t][r] * inv[r];
        }
    }
}

// ---------------------------------------------------------------------------
// Kernel 4: LayerNorm over D, in place on d_out. One block per (b,s) row.
// ---------------------------------------------------------------------------
__global__ __launch_bounds__(256)
void ln_kernel(float* __restrict__ out, const float* __restrict__ ln_scale)
{
    __shared__ float red[256];
    const int bs = blockIdx.x;
    float* row = out + (size_t)bs * D_;
    const int tid = threadIdx.x;

    float x[4];
    float sum = 0.0f;
#pragma unroll
    for (int i = 0; i < 4; ++i) { x[i] = row[tid + i * 256]; sum += x[i]; }
    red[tid] = sum;
    __syncthreads();
    for (int st = 128; st > 0; st >>= 1) {
        if (tid < st) red[tid] += red[tid + st];
        __syncthreads();
    }
    const float mean = red[0] * (1.0f / D_);
    __syncthreads();

    float sq = 0.0f;
#pragma unroll
    for (int i = 0; i < 4; ++i) { const float d = x[i] - mean; sq += d * d; }
    red[tid] = sq;
    __syncthreads();
    for (int st = 128; st > 0; st >>= 1) {
        if (tid < st) red[tid] += red[tid + st];
        __syncthreads();
    }
    const float var  = red[0] * (1.0f / D_);
    const float rstd = rsqrtf(var + LNEPS);
#pragma unroll
    for (int i = 0; i < 4; ++i) {
        const int d = tid + i * 256;
        row[d] = (x[i] - mean) * rstd * ln_scale[d];
    }
}

// ---------------------------------------------------------------------------
extern "C" void kernel_launch(void* const* d_in, const int* in_sizes, int n_in,
                              void* d_out, int out_size, void* d_ws, size_t ws_size,
                              hipStream_t stream)
{
    const float* q   = (const float*)d_in[0];
    const float* k   = (const float*)d_in[1];
    const float* v   = (const float*)d_in[2];
    const float* igw = (const float*)d_in[3];
    const float* igb = (const float*)d_in[4];
    const float* fgw = (const float*)d_in[5];
    const float* fgb = (const float*)d_in[6];
    const float* lns = (const float*)d_in[7];
    float* out = (float*)d_out;

    float* csbuf = (float*)d_ws;                          // log_f -> cs   [B*NH*S]
    float* abuf  = csbuf + (size_t)B_ * NH_ * S_;         // ig    -> a    [B*NH*S]

    gate_kernel<<<B_ * S_, 256, 0, stream>>>(q, k, v, igw, igb, fgw, fgb, abuf, csbuf);
    scan_kernel<<<B_ * NH_, 256, 0, stream>>>(csbuf, abuf);
    mlstm_attn_kernel<<<B_ * NH_ * (S_ / BM_), 256, 0, stream>>>(q, k, v, csbuf, abuf, out);
    ln_kernel<<<B_ * S_, 256, 0, stream>>>(out, lns);
}
